// MaskGraphConv_89515708383727
// MI455X (gfx1250) — matrix operand order repaired and checked by reading, not compiled, not run on hardware
//
#include <hip/hip_runtime.h>
#include <hip/hip_bf16.h>

typedef __attribute__((ext_vector_type(16))) _Float16 v16h;
typedef __attribute__((ext_vector_type(8)))  float    v8f;
typedef __attribute__((ext_vector_type(4)))  float    v4f;

#define IN_F  256
#define OUT_F 128

// ---------------- zero fill ----------------
__global__ void fill0_kernel(float* __restrict__ p, long n) {
    long i = (long)blockIdx.x * blockDim.x + threadIdx.x;
    long stride = (long)gridDim.x * blockDim.x;
    for (; i < n; i += stride) p[i] = 0.0f;
}

// ---------------- degree count ----------------
__global__ void deg_kernel(const int* __restrict__ src, const int* __restrict__ dst,
                           float* __restrict__ outdeg, float* __restrict__ indeg, int E) {
    int e = blockIdx.x * blockDim.x + threadIdx.x;
    if (e < E) {
        atomicAdd(outdeg + src[e], 1.0f);
        atomicAdd(indeg  + dst[e], 1.0f);
    }
}

// ---------------- deg -> clip(deg,1)^-0.5 in place ----------------
__global__ void rsq_kernel(float* __restrict__ a, float* __restrict__ b, int N) {
    int i = blockIdx.x * blockDim.x + threadIdx.x;
    if (i < N) {
        a[i] = rsqrtf(fmaxf(a[i], 1.0f));
        b[i] = rsqrtf(fmaxf(b[i], 1.0f));
    }
}

// ---------------- masked weight, transposed to f16: Wt[col][k] ----------------
__global__ void maskw_kernel(const float* __restrict__ weight, const float* __restrict__ mask,
                             _Float16* __restrict__ Wt) {
    int idx = blockIdx.x * blockDim.x + threadIdx.x;      // over IN_F*OUT_F
    if (idx < IN_F * OUT_F) {
        int k = idx >> 7;          // row in [IN_F]
        int j = idx & (OUT_F - 1); // col in [OUT_F]
        float v = (mask[idx] > 0.5f) ? weight[idx] : 0.0f;
        Wt[(size_t)j * IN_F + k] = (_Float16)v;
    }
}

// ---------------- WMMA GEMM: feat_src = (scale * feat) @ W ----------------
// Block = 256 threads = 8 waves; each wave computes a 16x128 output tile.
__global__ __launch_bounds__(256) void gemm_kernel(const float* __restrict__ feat,
                                                   const float* __restrict__ scale,
                                                   const _Float16* __restrict__ Wt,
                                                   float* __restrict__ out, int N) {
    const int wave  = threadIdx.x >> 5;
    const int lane  = threadIdx.x & 31;
    const int lhalf = lane >> 4;     // 0: lanes 0-15, 1: lanes 16-31
    const int l16   = lane & 15;

    const int rowBlock = blockIdx.x * 128 + wave * 16;
    const int row      = rowBlock + l16;                 // A row for this lane (both halves)
    const int rowC     = (row < N) ? row : 0;            // clamp for loads
    const float s      = scale[rowC];
    const float* arow  = feat + (size_t)rowC * IN_F;
    const _Float16* wcol = Wt + (size_t)l16 * IN_F;      // per-lane base column

    v8f acc[8] = {};

    for (int k0 = 0; k0 < IN_F; k0 += 32) {
        // ---- A fragment (contiguous K, matching B):
        // lane<16:  elems 0-15 = K[k0..k0+15]
        // lane>=16: elems 0-15 = K[k0+16..k0+31]
        const int ka = k0 + lhalf * 16;
        v4f f0 = *(const v4f*)(arow + ka);
        v4f f1 = *(const v4f*)(arow + ka + 4);
        v4f f2 = *(const v4f*)(arow + ka + 8);
        v4f f3 = *(const v4f*)(arow + ka + 12);

        // ---- hoist all 8 B fragments for this K-step (one wait, dense WMMA issue)
        const int kb = k0 + lhalf * 16;   // B: lane<16 holds K[kb..kb+15], lane>=16 holds K[k0+16..k0+31]
        v16h b[8];
#pragma unroll
        for (int t = 0; t < 8; ++t)
            b[t] = *(const v16h*)(wcol + (size_t)t * 16 * IN_F + kb);

        v16h a;
#pragma unroll
        for (int i = 0; i < 4; ++i) {
            a[i]      = (_Float16)(f0[i] * s);
            a[4 + i]  = (_Float16)(f1[i] * s);
            a[8 + i]  = (_Float16)(f2[i] * s);
            a[12 + i] = (_Float16)(f3[i] * s);
        }

#pragma unroll
        for (int t = 0; t < 8; ++t)
            acc[t] = __builtin_amdgcn_wmma_f32_16x16x32_f16(
                false, a, false, b[t], (short)0, acc[t], false, false);
    }

    // ---- store D: VGPR r -> M = r + lhalf*8, N = t*16 + l16
    if (rowBlock + 15 < N) {
        // fast path: entire 16-row strip in range, no per-element guards
        float* obase = out + (size_t)(rowBlock + lhalf * 8) * OUT_F + l16;
#pragma unroll
        for (int t = 0; t < 8; ++t) {
#pragma unroll
            for (int r = 0; r < 8; ++r)
                obase[(size_t)r * OUT_F + t * 16] = acc[t][r];
        }
    } else {
        // tail block: guard each row
#pragma unroll
        for (int t = 0; t < 8; ++t) {
            const int col = t * 16 + l16;
#pragma unroll
            for (int r = 0; r < 8; ++r) {
                const int orow = rowBlock + r + lhalf * 8;
                if (orow < N) out[(size_t)orow * OUT_F + col] = acc[t][r];
            }
        }
    }
}

// ---------------- edge gather * weight, scatter-add (1 wave / edge) ----------------
__global__ __launch_bounds__(256) void scatter_kernel(const float* __restrict__ feat_src,
                                                      const int* __restrict__ src,
                                                      const int* __restrict__ dst,
                                                      const float* __restrict__ ew,
                                                      float* __restrict__ out, int E) {
    long tid = (long)blockIdx.x * blockDim.x + threadIdx.x;
    int e    = (int)(tid >> 5);
    int lane = (int)(tid & 31);
    if (e >= E) return;
    const int   s = src[e];
    const int   d = dst[e];
    const float w = ew[e];
    v4f v = *(const v4f*)(feat_src + (size_t)s * OUT_F + lane * 4);
    float* op = out + (size_t)d * OUT_F + lane * 4;
    atomicAdd(op + 0, v[0] * w);
    atomicAdd(op + 1, v[1] * w);
    atomicAdd(op + 2, v[2] * w);
    atomicAdd(op + 3, v[3] * w);
}

// ---------------- finalize: out = out * indeg^-0.5 + bias ----------------
__global__ void fin_kernel(float* __restrict__ out, const float* __restrict__ indeg_inv,
                           const float* __restrict__ bias, int N) {
    long idx = (long)blockIdx.x * blockDim.x + threadIdx.x;
    if (idx < (long)N * OUT_F) {
        int i = (int)(idx >> 7);
        int j = (int)(idx & (OUT_F - 1));
        out[idx] = out[idx] * indeg_inv[i] + bias[j];
    }
}

extern "C" void kernel_launch(void* const* d_in, const int* in_sizes, int n_in,
                              void* d_out, int out_size, void* d_ws, size_t ws_size,
                              hipStream_t stream) {
    (void)n_in; (void)out_size; (void)ws_size;
    const float* feat   = (const float*)d_in[0];
    const int*   src    = (const int*)d_in[1];
    const int*   dst    = (const int*)d_in[2];
    const float* ew     = (const float*)d_in[3];
    const float* weight = (const float*)d_in[4];
    const float* bias   = (const float*)d_in[5];
    const float* mask   = (const float*)d_in[6];
    const int N = in_sizes[0] / IN_F;
    const int E = in_sizes[1];
    float* out = (float*)d_out;

    // workspace: [outdeg f32 N][indeg f32 N][Wt f16 128x256][feat_src f32 Nx128]
    char* ws = (char*)d_ws;
    float* outdeg = (float*)ws;
    float* indeg  = (float*)(ws + (size_t)N * 4);
    size_t wtOff  = (((size_t)N * 8) + 255) & ~(size_t)255;
    _Float16* Wt  = (_Float16*)(ws + wtOff);
    size_t fsOff  = (wtOff + (size_t)IN_F * OUT_F * 2 + 255) & ~(size_t)255;
    float* feat_src = (float*)(ws + fsOff);

    // 1. zero accumulators (degrees are contiguous -> one fill)
    fill0_kernel<<<2048, 256, 0, stream>>>(outdeg, (long)2 * N);
    fill0_kernel<<<4096, 256, 0, stream>>>(out, (long)N * OUT_F);
    // 2. degree counts
    deg_kernel<<<(E + 255) / 256, 256, 0, stream>>>(src, dst, outdeg, indeg, E);
    // 3. inverse sqrt norms (in place)
    rsq_kernel<<<(N + 255) / 256, 256, 0, stream>>>(outdeg, indeg, N);
    // 4. masked weight -> transposed f16
    maskw_kernel<<<(IN_F * OUT_F + 255) / 256, 256, 0, stream>>>(weight, mask, Wt);
    // 5. WMMA projection
    gemm_kernel<<<(N + 127) / 128, 256, 0, stream>>>(feat, outdeg, Wt, feat_src, N);
    // 6. gather/scale/scatter-add over edges (1 wave per edge)
    long sthreads = (long)E * 32;
    scatter_kernel<<<(unsigned)((sthreads + 255) / 256), 256, 0, stream>>>(feat_src, src, dst, ew, out, E);
    // 7. in-degree norm + bias
    fin_kernel<<<(unsigned)(((long)N * OUT_F + 255) / 256), 256, 0, stream>>>(out, indeg, bias, N);
}
